// SOBOG_386547057103
// MI455X (gfx1250) — matrix operand order repaired and checked by reading, not compiled, not run on hardware
//
#include <hip/hip_runtime.h>
#include <hip/hip_bf16.h>

#define NN    8192
#define EE    65536
#define HH    2
#define CC    256
#define HC    512
#define NHOPS 5
#define NLAY  2

typedef __attribute__((ext_vector_type(16))) __bf16 v16bf;
typedef __attribute__((ext_vector_type(8)))  float  v8f;

// ---------------------------------------------------------------------------
// Weight prep: Bt[n*K + k] = bf16(B[k*Nc + n])  (transpose + f32->bf16)
// Makes the WMMA B-operand a contiguous 32B-aligned v16bf load per lane.
// ---------------------------------------------------------------------------
__global__ void conv_bt(const float* __restrict__ B, __bf16* __restrict__ Bt,
                        int K, int Nc)
{
  int t = blockIdx.x * blockDim.x + threadIdx.x;
  if (t >= K * Nc) return;
  int k = t / Nc, n = t - k * Nc;
  Bt[(size_t)n * K + k] = (__bf16)B[t];
}

// ---------------------------------------------------------------------------
// bf16-WMMA GEMM:  Cmat[M,Nc] = A[M,K] @ B[K,Nc] (+ bias[Nc])
// A fp32 row-major (lda==K), B pre-transposed bf16 (Bt[Nc][K]), C fp32 (ldc==Nc).
// One wave32 computes a 16x32 output tile: A converted once, 2 WMMAs per K-step.
// ---------------------------------------------------------------------------
__global__ __launch_bounds__(256) void wmma_gemm_bf16(
    const float* __restrict__ A, const __bf16* __restrict__ Bt,
    const float* __restrict__ bias, float* __restrict__ Cmat,
    int M, int K, int Nc)
{
  const int lane = threadIdx.x & 31;
  const int wave = threadIdx.x >> 5;
  const int tiles_n = Nc >> 5;               // 32-wide tile pairs
  const int tile = blockIdx.x * (blockDim.x >> 5) + wave;
  if (tile >= (M >> 4) * tiles_n) return;
  const int tm = tile / tiles_n, tn = tile % tiles_n;
  const int half = lane >> 4;                // 0 or 1
  const int l16  = lane & 15;

  const float*  arow = A  + (size_t)(tm * 16 + l16) * K;
  const __bf16* b0   = Bt + (size_t)(tn * 32 + l16) * K + half * 16;
  const __bf16* b1   = b0 + (size_t)16 * K;

  v8f acc0 = {0.f, 0.f, 0.f, 0.f, 0.f, 0.f, 0.f, 0.f};
  v8f acc1 = {0.f, 0.f, 0.f, 0.f, 0.f, 0.f, 0.f, 0.f};

#pragma unroll 4
  for (int kb = 0; kb < K; kb += 32) {
    // A 16x32 bf16 layout: element e -> K = kb + 16*half + e
    v16bf va;
    const float* a0 = arow + kb + half * 16;
    const float* a1 = a0 + 8;
#pragma unroll
    for (int e = 0; e < 8; ++e) {
      va[e]     = (__bf16)a0[e];
      va[e + 8] = (__bf16)a1[e];
    }
    // B 32x16 bf16 layout: element e -> K = kb + 16*half + e, column = l16
    v16bf vb0 = *(const v16bf*)(b0 + kb);    // 32B aligned (K,kb mult of 32)
    v16bf vb1 = *(const v16bf*)(b1 + kb);

    acc0 = __builtin_amdgcn_wmma_f32_16x16x32_bf16(
        false, va, false, vb0, (short)0, acc0, false, false);
    acc1 = __builtin_amdgcn_wmma_f32_16x16x32_bf16(
        false, va, false, vb1, (short)0, acc1, false, false);
  }

  const int col0 = tn * 32 + l16;
  const int col1 = col0 + 16;
  const float bv0 = bias ? bias[col0] : 0.f;
  const float bv1 = bias ? bias[col1] : 0.f;
#pragma unroll
  for (int r = 0; r < 8; ++r) {
    const int row = tm * 16 + r + half * 8;  // C/D layout: VGPR r, half -> M+8
    Cmat[(size_t)row * Nc + col0] = acc0[r] + bv0;
    Cmat[(size_t)row * Nc + col1] = acc1[r] + bv1;
  }
}

// ---------------------------------------------------------------------------
__global__ void fill_kernel(float* __restrict__ p, float v, int n) {
  int t = blockIdx.x * blockDim.x + threadIdx.x;
  if (t < n) p[t] = v;
}

__device__ inline void atomicMaxFloat(float* addr, float v) {
  if (v >= 0.f) atomicMax((int*)addr, __float_as_int(v));
  else          atomicMin((unsigned int*)addr, __float_as_uint(v));
}

// es[t]= x[t,:]·a_src[h]; ed[t]= x[t,:]·a_dst[h];  t = n*H + h
__global__ void att_scores(const float* __restrict__ x,
                           const float* __restrict__ a_src,
                           const float* __restrict__ a_dst,
                           float* __restrict__ es, float* __restrict__ ed, int NH)
{
  int t = blockIdx.x * blockDim.x + threadIdx.x;
  if (t >= NH) return;
  int h = t & (HH - 1);
  const float* xv = x + (size_t)t * CC;
  const float* as = a_src + h * CC;
  const float* ad = a_dst + h * CC;
  float s = 0.f, d = 0.f;
  for (int c = 0; c < CC; ++c) { float v = xv[c]; s += v * as[c]; d += v * ad[c]; }
  es[t] = s; ed[t] = d;
}

__device__ inline float edge_val(const float* es, const float* ed,
                                 const int* src, const int* dst, int e, int h) {
  float v = es[src[e] * HH + h] + ed[dst[e] * HH + h];
  return v > 0.f ? v : 0.2f * v;           // leaky_relu 0.2
}

__global__ void edge_max(const float* __restrict__ es, const float* __restrict__ ed,
                         const int* __restrict__ src, const int* __restrict__ dst,
                         float* __restrict__ emax, int EHt)
{
  int t = blockIdx.x * blockDim.x + threadIdx.x;
  if (t >= EHt) return;
  int e = t >> 1, h = t & 1;
  atomicMaxFloat(&emax[dst[e] * HH + h], edge_val(es, ed, src, dst, e, h));
}

__global__ void edge_exp(const float* __restrict__ es, const float* __restrict__ ed,
                         const int* __restrict__ src, const int* __restrict__ dst,
                         const float* __restrict__ emax,
                         float* __restrict__ ee, float* __restrict__ den, int EHt)
{
  int t = blockIdx.x * blockDim.x + threadIdx.x;
  if (t >= EHt) return;
  int e = t >> 1, h = t & 1;
  float v = edge_val(es, ed, src, dst, e, h);
  float ex = __expf(v - emax[dst[e] * HH + h]);
  ee[t] = ex;
  atomicAdd(&den[dst[e] * HH + h], ex);
}

// y[dst,h,:] += alpha * x[src,h,:]   (wave32 per (edge,head), 8 channels/lane)
__global__ __launch_bounds__(256) void scatter_accum(
    const float* __restrict__ x, const float* __restrict__ ee,
    const float* __restrict__ den, const int* __restrict__ src,
    const int* __restrict__ dst, float* __restrict__ y, int EHt)
{
  int wid = (blockIdx.x * blockDim.x + threadIdx.x) >> 5;
  if (wid >= EHt) return;
  int lane = threadIdx.x & 31;
  int e = wid >> 1, h = wid & 1;
  int s = src[e], d = dst[e];
  // prefetch a future edge's source row (gfx1250 global_prefetch_b8 path)
  int wnext = wid + 256;
  if (wnext < EHt)
    __builtin_prefetch(x + ((size_t)src[wnext >> 1] * HH + (wnext & 1)) * CC, 0, 1);

  float alpha = ee[wid] / (den[d * HH + h] + 1e-16f);
  const float4* xs = (const float4*)(x + ((size_t)s * HH + h) * CC) + lane * 2;
  float*        yd = y + ((size_t)d * HH + h) * CC + lane * 8;
  float4 v0 = xs[0], v1 = xs[1];
  atomicAdd(&yd[0], alpha * v0.x); atomicAdd(&yd[1], alpha * v0.y);
  atomicAdd(&yd[2], alpha * v0.z); atomicAdd(&yd[3], alpha * v0.w);
  atomicAdd(&yd[4], alpha * v1.x); atomicAdd(&yd[5], alpha * v1.y);
  atomicAdd(&yd[6], alpha * v1.z); atomicAdd(&yd[7], alpha * v1.w);
}

// pe[n,c] = mean over heads + bias
__global__ void mean_bias(const float* __restrict__ x, const float* __restrict__ bias,
                          float* __restrict__ pe)
{
  int t = blockIdx.x * blockDim.x + threadIdx.x;   // t = n*CC + c
  int n = t >> 8, c = t & (CC - 1);
  pe[t] = 0.5f * (x[((size_t)n * HH) * CC + c] + x[((size_t)n * HH + 1) * CC + c]) + bias[c];
}

// column-wise max over 64 rows per block, folded with atomicMaxFloat
__global__ __launch_bounds__(256) void rowmax(const float* __restrict__ pe,
                                              float* __restrict__ agg)
{
  int c = threadIdx.x;
  int r0 = blockIdx.x * 64;
  float m = -__builtin_huge_valf();
  for (int r = 0; r < 64; ++r) {
    float v = pe[(size_t)(r0 + r) * CC + c];
    m = v > m ? v : m;
  }
  atomicMaxFloat(&agg[c], m);
}

__global__ void ue_kernel(const float* __restrict__ users, const float* __restrict__ uw,
                          const float* __restrict__ ub, float* __restrict__ ue)
{
  int j = threadIdx.x;                 // 256 threads
  float acc = ub[j];
  for (int i = 0; i < 128; ++i) acc += users[i] * uw[i * 256 + j];
  ue[j] = acc;
}

__global__ __launch_bounds__(512) void user_mlp(
    const float* __restrict__ ue, const float* __restrict__ agg,
    const float* __restrict__ w0, const float* __restrict__ b0,
    const float* __restrict__ w1, const float* __restrict__ b1,
    const float* __restrict__ w2, const float* __restrict__ b2,
    float* __restrict__ out0)
{
  __shared__ float up[512], h1[512], red[512];
  int t = threadIdx.x;
  up[t] = (t < 256) ? ue[t] : agg[t - 256];
  __syncthreads();
  float acc = b0[t];
  for (int i = 0; i < 512; ++i) acc += up[i] * w0[i * 512 + t];
  h1[t] = acc;
  __syncthreads();
  acc = b1[t];
  for (int i = 0; i < 512; ++i) acc += h1[i] * w1[i * 512 + t];
  red[t] = acc * w2[t];
  __syncthreads();
  for (int s = 256; s > 0; s >>= 1) {
    if (t < s) red[t] += red[t + s];
    __syncthreads();
  }
  if (t == 0) out0[0] = 1.f / (1.f + __expf(-(red[0] + b2[0])));
}

__global__ void post_logits(const float* __restrict__ z, const float* __restrict__ w2,
                            const float* __restrict__ b2, float* __restrict__ out, int Nn)
{
  int n = blockIdx.x * blockDim.x + threadIdx.x;
  if (n >= Nn) return;
  const float4* zr = (const float4*)(z + (size_t)n * CC);
  const float4* w4 = (const float4*)w2;
  float acc = b2[0];
  for (int i = 0; i < CC / 4; ++i) {
    float4 a = zr[i], b = w4[i];
    acc += a.x * b.x + a.y * b.y + a.z * b.z + a.w * b.w;
  }
  out[n] = 1.f / (1.f + __expf(-acc));
}

// ---------------------------------------------------------------------------
extern "C" void kernel_launch(void* const* d_in, const int* in_sizes, int n_in,
                              void* d_out, int out_size, void* d_ws, size_t ws_size,
                              hipStream_t stream)
{
  const float* users = (const float*)d_in[0];
  const float* posts = (const float*)d_in[1];
  const float* uw  = (const float*)d_in[2];  const float* ub  = (const float*)d_in[3];
  const float* pw  = (const float*)d_in[4];  const float* pb  = (const float*)d_in[5];
  const float* gat_lin = (const float*)d_in[6];
  const float* gat_asrc = (const float*)d_in[7];
  const float* gat_adst = (const float*)d_in[8];
  const float* gat_bias = (const float*)d_in[9];
  const float* pcw0 = (const float*)d_in[10]; const float* pcb0 = (const float*)d_in[11];
  const float* pcw1 = (const float*)d_in[12]; const float* pcb1 = (const float*)d_in[13];
  const float* pcw2 = (const float*)d_in[14]; const float* pcb2 = (const float*)d_in[15];
  const float* ucw0 = (const float*)d_in[16]; const float* ucb0 = (const float*)d_in[17];
  const float* ucw1 = (const float*)d_in[18]; const float* ucb1 = (const float*)d_in[19];
  const float* ucw2 = (const float*)d_in[20]; const float* ucb2 = (const float*)d_in[21];
  const int*   eidx = (const int*)d_in[22];
  const int* src = eidx;
  const int* dst = eidx + EE;

  float* ws   = (float*)d_ws;
  float* pe   = ws;  ws += (size_t)NN * CC;        // 8 MB
  float* xbuf = ws;  ws += (size_t)NN * HH * CC;   // 16 MB
  float* ybuf = ws;  ws += (size_t)NN * HH * CC;   // 16 MB
  float* es   = ws;  ws += NN * HH;
  float* ed   = ws;  ws += NN * HH;
  float* emax = ws;  ws += NN * HH;
  float* den  = ws;  ws += NN * HH;
  float* ee   = ws;  ws += (size_t)EE * HH;
  float* uebf = ws;  ws += 256;
  float* agg  = ws;  ws += 256;
  __bf16* btw = (__bf16*)ws;  ws += (512 * 512) / 2;   // 512KB bf16 staging
  float* out  = (float*)d_out;

  const float NEGINF = -__builtin_huge_valf();
  const int NH = NN * HH, EH = EE * HH;

  // posts embed: pe = posts @ pw + pb   [8192,512]x[512,256]
  conv_bt<<<(512 * CC) / 256, 256, 0, stream>>>(pw, btw, 512, CC);
  wmma_gemm_bf16<<<(NN / 16) * (CC / 32) / 8, 256, 0, stream>>>(
      posts, btw, pb, pe, NN, 512, CC);

  float* x = xbuf; float* y = ybuf;
  for (int l = 0; l < NLAY; ++l) {
    // x = pe @ lin[l]   [8192,256]x[256,512]
    conv_bt<<<(CC * HC) / 256, 256, 0, stream>>>(gat_lin + (size_t)l * CC * HC,
                                                 btw, CC, HC);
    wmma_gemm_bf16<<<(NN / 16) * (HC / 32) / 8, 256, 0, stream>>>(
        pe, btw, nullptr, x, NN, CC, HC);
    for (int hop = 0; hop < NHOPS; ++hop) {
      fill_kernel<<<(NN * HH * CC) / 256, 256, 0, stream>>>(y, 0.f, NN * HH * CC);
      fill_kernel<<<(NH + 255) / 256, 256, 0, stream>>>(emax, NEGINF, NH);
      fill_kernel<<<(NH + 255) / 256, 256, 0, stream>>>(den, 0.f, NH);
      att_scores<<<NH / 256, 256, 0, stream>>>(
          x, gat_asrc + l * HH * CC, gat_adst + l * HH * CC, es, ed, NH);
      edge_max<<<EH / 256, 256, 0, stream>>>(es, ed, src, dst, emax, EH);
      edge_exp<<<EH / 256, 256, 0, stream>>>(es, ed, src, dst, emax, ee, den, EH);
      scatter_accum<<<EH * 32 / 256, 256, 0, stream>>>(x, ee, den, src, dst, y, EH);
      float* t = x; x = y; y = t;
    }
    mean_bias<<<(NN * CC) / 256, 256, 0, stream>>>(x, gat_bias + l * CC, pe);
  }

  // user head
  fill_kernel<<<1, 256, 0, stream>>>(agg, NEGINF, 256);
  rowmax<<<NN / 64, 256, 0, stream>>>(pe, agg);
  ue_kernel<<<1, 256, 0, stream>>>(users, uw, ub, uebf);
  user_mlp<<<1, 512, 0, stream>>>(uebf, agg, ucw0, ucb0, ucw1, ucb1, ucw2, ucb2, out);

  // post classifier (reuse x/y buffers as z0/z1)
  float* z0 = x; float* z1 = y;
  conv_bt<<<(CC * CC) / 256, 256, 0, stream>>>(pcw0, btw, CC, CC);
  wmma_gemm_bf16<<<(NN / 16) * (CC / 32) / 8, 256, 0, stream>>>(
      pe, btw, pcb0, z0, NN, CC, CC);
  conv_bt<<<(CC * CC) / 256, 256, 0, stream>>>(pcw1, btw, CC, CC);
  wmma_gemm_bf16<<<(NN / 16) * (CC / 32) / 8, 256, 0, stream>>>(
      z0, btw, pcb1, z1, NN, CC, CC);
  post_logits<<<NN / 256, 256, 0, stream>>>(z1, pcw2, pcb2, out + 1, NN);
}
